// TopDownAttention_75608604279276
// MI455X (gfx1250) — matrix operand order repaired and checked by reading, not compiled, not run on hardware
//
#include <hip/hip_runtime.h>

// ---------------------------------------------------------------------------
// TopDownAttention for MI455X (gfx1250, wave32, WMMA)
// B=256, O=100, F_IMG=2048, F_Q=1024, N_ATTN=512, PH=PW=14
// ---------------------------------------------------------------------------

#define Bsz   256
#define Osz   100
#define FIMG  2048
#define FQ    1024
#define DIN   3072            // FIMG + FQ
#define NATT  512
#define PIX   196             // 14*14
#define MROWS (Bsz * Osz)     // 25600

typedef __attribute__((ext_vector_type(16))) __bf16 bf16x16;
typedef __attribute__((ext_vector_type(8)))  float  f32x8;

// ---------------------------------------------------------------------------
// Kernel 0: weight-norm both linear layers.
//   blocks 0..511 : Wn[n, :] = bf16( gate_g[n] * gate_v[n,:] / ||gate_v[n,:]|| )
//   block  512    : Wa[:]    = attn_g * attn_v / ||attn_v||   (fp32)
// ---------------------------------------------------------------------------
__global__ __launch_bounds__(256) void tda_prep_weights(
    const float* __restrict__ gate_v, const float* __restrict__ gate_g,
    const float* __restrict__ attn_v, const float* __restrict__ attn_g,
    __bf16* __restrict__ Wn, float* __restrict__ Wa)
{
    __shared__ float red[256];
    const int tid = threadIdx.x;

    if (blockIdx.x < NATT) {
        const int n = blockIdx.x;
        const float* v = gate_v + (size_t)n * DIN;
        float ss = 0.f;
        for (int k = tid; k < DIN; k += 256) { float x = v[k]; ss += x * x; }
        red[tid] = ss; __syncthreads();
        for (int off = 128; off > 0; off >>= 1) {
            if (tid < off) red[tid] += red[tid + off];
            __syncthreads();
        }
        const float scale = gate_g[n] / sqrtf(red[0]);
        for (int k = tid; k < DIN; k += 256)
            Wn[(size_t)n * DIN + k] = (__bf16)(v[k] * scale);
    } else {
        float ss = 0.f;
        for (int k = tid; k < NATT; k += 256) { float x = attn_v[k]; ss += x * x; }
        red[tid] = ss; __syncthreads();
        for (int off = 128; off > 0; off >>= 1) {
            if (tid < off) red[tid] += red[tid + off];
            __syncthreads();
        }
        const float scale = attn_g[0] / sqrtf(red[0]);
        for (int k = tid; k < NATT; k += 256)
            Wa[k] = attn_v[k] * scale;
    }
}

// ---------------------------------------------------------------------------
// Kernel 1: fused score GEMM.
//   score[m] = attn_b + sum_n  Wa[n] * relu( x[m,:] . Wn[n,:] + gate_b[n] )
//   x[m,:] = concat( obj_feats[m,:2048], ques_emb[m/O, :1024] )
//
// One wave owns a 32-row tile of M (two 16x32 A fragments), so every B
// fragment feeds TWO v_wmma_f32_16x16x32_bf16 ops (16 WMMAs per 24 b128
// loads per K-step).  N processed in 4 chunks of 128 (2x8 accumulators).
//
// Fragment layouts per CDNA5 ISA 7.12.2 (16-bit, 16x32 A / 32x16 B):
//   A: lane l -> row (l&15); elems 0..7 = K[koff..koff+7],
//      elems 8..15 = K[16+koff .. 23+koff], koff = 8*(l>>4)
//   B: lane l -> col (l&15); elems in the same K order as A (koff, 16+koff)
//   C/D: lane l -> col (l&15); vgpr r -> row r + 8*(l>>4)
// ---------------------------------------------------------------------------
__global__ __launch_bounds__(128) void tda_score_gemm(
    const float*  __restrict__ obj_feats,
    const float*  __restrict__ ques_emb,
    const __bf16* __restrict__ Wn,
    const float*  __restrict__ Wa,
    const float*  __restrict__ gate_b,
    const float*  __restrict__ attn_b,
    float* __restrict__ score)
{
    const int lane = threadIdx.x & 31;
    const int wv   = threadIdx.x >> 5;           // wave in block (0..3)
    const int m0   = blockIdx.x * 128 + wv * 32; // 32-row tile base
    const int half = lane >> 4;                  // 0 or 1
    const int ncol = lane & 15;

    const int arow0 = m0 + ncol;                 // A rows for this lane
    const int arow1 = m0 + 16 + ncol;
    const int abat0 = arow0 / Osz;               // batches (for ques portion)
    const int abat1 = arow1 / Osz;
    const int koff  = half << 3;                 // A half K-offset: 0 / 8
    const int bko   = half << 4;                 // unused; B loads use koff like A

    float sacc0[8], sacc1[8];
    #pragma unroll
    for (int r = 0; r < 8; ++r) { sacc0[r] = 0.f; sacc1[r] = 0.f; }

    const float ab0 = attn_b[0];

    for (int nc = 0; nc < 4; ++nc) {             // 4 chunks of 128 N-columns
        f32x8 acc0[8], acc1[8];
        #pragma unroll
        for (int t = 0; t < 8; ++t)
            #pragma unroll
            for (int r = 0; r < 8; ++r) { acc0[t][r] = 0.f; acc1[t][r] = 0.f; }

        for (int kt = 0; kt < DIN / 32; ++kt) {  // 96 K-steps of 32
            const int k0 = kt * 32;
            // K block never straddles the 2048 boundary (2048 % 32 == 0)
            const bool img = (k0 < FIMG);
            const float* base0 = img
                ? obj_feats + (size_t)arow0 * FIMG + k0
                : ques_emb + (size_t)abat0 * FQ + (k0 - FIMG);
            const float* base1 = img
                ? obj_feats + (size_t)arow1 * FIMG + k0
                : ques_emb + (size_t)abat1 * FQ + (k0 - FIMG);

            float fv0[16], fv1[16];
            *(float4*)&fv0[0]  = *(const float4*)(base0 + koff);
            *(float4*)&fv0[4]  = *(const float4*)(base0 + koff + 4);
            *(float4*)&fv0[8]  = *(const float4*)(base0 + 16 + koff);
            *(float4*)&fv0[12] = *(const float4*)(base0 + 16 + koff + 4);
            *(float4*)&fv1[0]  = *(const float4*)(base1 + koff);
            *(float4*)&fv1[4]  = *(const float4*)(base1 + koff + 4);
            *(float4*)&fv1[8]  = *(const float4*)(base1 + 16 + koff);
            *(float4*)&fv1[12] = *(const float4*)(base1 + 16 + koff + 4);

            bf16x16 afrag0, afrag1;
            #pragma unroll
            for (int i = 0; i < 16; ++i) {
                afrag0[i] = (__bf16)fv0[i];
                afrag1[i] = (__bf16)fv1[i];
            }

            #pragma unroll
            for (int t = 0; t < 8; ++t) {
                const int n = (nc * 8 + t) * 16 + ncol;
                typedef __attribute__((ext_vector_type(8))) __bf16 bf16x8_t;
                const bf16x8_t blo = *(const bf16x8_t*)(Wn + (size_t)n * DIN + k0 + koff);
                const bf16x8_t bhi = *(const bf16x8_t*)(Wn + (size_t)n * DIN + k0 + 16 + koff);
                const bf16x16 bfrag = __builtin_shufflevector(blo, bhi, 0,1,2,3,4,5,6,7,8,9,10,11,12,13,14,15);
                acc0[t] = __builtin_amdgcn_wmma_f32_16x16x32_bf16(
                    false, afrag0, false, bfrag, (short)0, acc0[t], false, false);
                acc1[t] = __builtin_amdgcn_wmma_f32_16x16x32_bf16(
                    false, afrag1, false, bfrag, (short)0, acc1[t], false, false);
            }
        }

        // bias + relu + scale-by-Wa, fold into per-row partials
        #pragma unroll
        for (int t = 0; t < 8; ++t) {
            const int n   = (nc * 8 + t) * 16 + ncol;
            const float bias = gate_b[n];
            const float wa   = Wa[n];
            #pragma unroll
            for (int r = 0; r < 8; ++r) {
                float g0 = acc0[t][r] + bias;
                float g1 = acc1[t][r] + bias;
                g0 = g0 > 0.f ? g0 : 0.f;
                g1 = g1 > 0.f ? g1 : 0.f;
                sacc0[r] = fmaf(g0, wa, sacc0[r]);
                sacc1[r] = fmaf(g1, wa, sacc1[r]);
            }
        }
    }

    // reduce over the 16 lanes of each half (each lane held a disjoint set of n)
    #pragma unroll
    for (int m = 1; m <= 8; m <<= 1)
        #pragma unroll
        for (int r = 0; r < 8; ++r) {
            sacc0[r] += __shfl_xor(sacc0[r], m, 32);
            sacc1[r] += __shfl_xor(sacc1[r], m, 32);
        }

    if (ncol == 0) {
        const int mbase = m0 + (half << 3);      // rows 0..7 / 8..15 of the tile
        #pragma unroll
        for (int r = 0; r < 8; ++r) {
            score[mbase + r]      = sacc0[r] + ab0;
            score[mbase + 16 + r] = sacc1[r] + ab0;
        }
    }
}

// ---------------------------------------------------------------------------
// Kernel 2: per-batch masked softmax + weighted reductions.
// ---------------------------------------------------------------------------
__global__ __launch_bounds__(256) void tda_softmax_out(
    const float* __restrict__ score,
    const int*   __restrict__ num_obj,
    const float* __restrict__ obj_feats,
    const float* __restrict__ region,
    float* __restrict__ out_feat,   // [B, FIMG]
    float* __restrict__ out_mask,   // [B, PIX]
    float* __restrict__ out_wt)     // [B, O]
{
    __shared__ float red[256];
    __shared__ float w[Osz];
    const int b   = blockIdx.x;
    const int tid = threadIdx.x;
    const int nv  = num_obj[b];

    const bool valid = (tid < Osz) && (tid < nv);
    float s = valid ? score[b * Osz + tid] : -INFINITY;

    red[tid] = s; __syncthreads();
    for (int off = 128; off > 0; off >>= 1) {
        if (tid < off) red[tid] = fmaxf(red[tid], red[tid + off]);
        __syncthreads();
    }
    const float mx = red[0]; __syncthreads();

    const float e = valid ? __expf(s - mx) : 0.f;
    red[tid] = e; __syncthreads();
    for (int off = 128; off > 0; off >>= 1) {
        if (tid < off) red[tid] += red[tid + off];
        __syncthreads();
    }
    const float inv = 1.f / red[0];

    if (tid < Osz) {
        const float wt = e * inv;
        w[tid] = wt;
        out_wt[b * Osz + tid] = wt;
    }
    __syncthreads();

    const float* of = obj_feats + (size_t)b * Osz * FIMG;
    for (int f = tid; f < FIMG; f += 256) {
        float acc = 0.f;
        #pragma unroll 4
        for (int o = 0; o < Osz; ++o)
            acc = fmaf(w[o], of[(size_t)o * FIMG + f], acc);
        out_feat[(size_t)b * FIMG + f] = acc;
    }

    const float* mf = region + (size_t)b * Osz * PIX;
    for (int p = tid; p < PIX; p += 256) {
        float acc = 0.f;
        #pragma unroll 4
        for (int o = 0; o < Osz; ++o)
            acc = fmaf(w[o], mf[o * PIX + p], acc);
        out_mask[(size_t)b * PIX + p] = acc;
    }
}

// ---------------------------------------------------------------------------
// Host launcher
// ---------------------------------------------------------------------------
extern "C" void kernel_launch(void* const* d_in, const int* in_sizes, int n_in,
                              void* d_out, int out_size, void* d_ws, size_t ws_size,
                              hipStream_t stream)
{
    const float* obj_feats = (const float*)d_in[0];
    const float* ques_emb  = (const float*)d_in[1];
    const int*   num_obj   = (const int*)  d_in[2];
    const float* region    = (const float*)d_in[3];
    const float* gate_v    = (const float*)d_in[4];
    const float* gate_g    = (const float*)d_in[5];
    const float* gate_b    = (const float*)d_in[6];
    const float* attn_v    = (const float*)d_in[7];
    const float* attn_g    = (const float*)d_in[8];
    const float* attn_b    = (const float*)d_in[9];

    char* ws = (char*)d_ws;
    __bf16* Wn    = (__bf16*)ws;                                      // 512*3072*2 B
    float*  Wa    = (float*)(ws + (size_t)NATT * DIN * 2);            // 512*4 B
    float*  score = (float*)(ws + (size_t)NATT * DIN * 2 + NATT * 4); // 25600*4 B

    float* out      = (float*)d_out;
    float* out_feat = out;                                  // [B, FIMG]
    float* out_mask = out + (size_t)Bsz * FIMG;             // [B, PIX]
    float* out_wt   = out + (size_t)Bsz * FIMG + (size_t)Bsz * PIX; // [B, O]

    tda_prep_weights<<<NATT + 1, 256, 0, stream>>>(
        gate_v, gate_g, attn_v, attn_g, Wn, Wa);

    tda_score_gemm<<<MROWS / 128, 128, 0, stream>>>(
        obj_feats, ques_emb, Wn, Wa, gate_b, attn_b, score);

    tda_softmax_out<<<Bsz, 256, 0, stream>>>(
        score, num_obj, obj_feats, region, out_feat, out_mask, out_wt);
}
